// Attention_NMT_20615843021385
// MI455X (gfx1250) — compile-verified, matrix operand order checked by reading, not run on hardware
//
#include <hip/hip_runtime.h>
#include <hip/hip_bf16.h>

// ---------------------------------------------------------------------------
// Problem constants (from reference)
// ---------------------------------------------------------------------------
static constexpr int Bz = 32;     // batch
static constexpr int Sl = 100;    // source length
static constexpr int Tl = 100;    // target length
static constexpr int E  = 512;    // embedding
static constexpr int H  = 512;    // lstm hidden
static constexpr int VT = 32000;  // target vocab
static constexpr int G4H = 4 * H;      // 2048 gates
static constexpr int H2  = 2 * H;      // 1024
static constexpr int H3  = 3 * H;      // 1536
static constexpr int CIN = E + 3 * H;  // 2048 classifier input

typedef __bf16 bf16_t;
typedef __attribute__((ext_vector_type(16))) __bf16 v16bf;
typedef __attribute__((ext_vector_type(8)))  __bf16 v8bf;
typedef __attribute__((ext_vector_type(8)))  float  v8f;

// ---------------------------------------------------------------------------
// WMMA fragment loader: 16x32 bf16 tile, row-major source with leading dim ld.
// Lane layout per CDNA5 ISA 7.12.2 (16-bit A-matrix 16x32):
//   lanes 0-15  (row = lane):    K = base+0..7  and base+16..23
//   lanes 16-31 (row = lane-16): K = base+8..15 and base+24..31
// B operand (W[N,K] row-major, row = output column n) uses the mirrored layout.
// ---------------------------------------------------------------------------
__device__ __forceinline__ v16bf load_frag16x32(const bf16_t* __restrict__ base, int ld) {
    int lane = threadIdx.x & 31;
    int r    = lane & 15;
    int hi   = lane >> 4;           // 0 or 1
    const bf16_t* p = base + (size_t)r * ld + hi * 8;
    v8bf lo = *(const v8bf*)(p);        // K base+0..7   (or +8..15)
    v8bf hh = *(const v8bf*)(p + 16);   // K base+16..23 (or +24..31)
    return __builtin_shufflevector(lo, hh, 0,1,2,3,4,5,6,7,8,9,10,11,12,13,14,15);
}

__device__ __forceinline__ v8f wmma_bf16(v16bf a, v16bf b, v8f c) {
    return __builtin_amdgcn_wmma_f32_16x16x32_bf16(false, a, false, b, (short)0, c, false, false);
}

// Epilogue for one 16x16 f32 tile. C/D layout: lane 0-15 col=lane rows 0..7;
// lane 16-31 rows 8..15.
__device__ __forceinline__ void store_tile(const v8f& acc, int mrow, int ncol,
                                           float* __restrict__ C, bf16_t* __restrict__ Cbf,
                                           int ldc, const float* __restrict__ addend,
                                           const float* __restrict__ bias, int relu)
{
    int lane    = threadIdx.x & 31;
    int col     = ncol + (lane & 15);
    int rowbase = mrow + (lane >> 4) * 8;
    float bv = bias ? bias[col] : 0.0f;
    #pragma unroll
    for (int r = 0; r < 8; ++r) {
        size_t idx = (size_t)(rowbase + r) * ldc + col;
        float  v   = acc[r] + bv;
        if (addend) v += addend[idx];
        if (relu && v < 0.0f) v = 0.0f;
        if (C)   C[idx]   = v;
        if (Cbf) Cbf[idx] = (bf16_t)v;
    }
}

// ---------------------------------------------------------------------------
// WMMA GEMM:  Out[M,N] = act( addend + A[M,K] @ W[N,K]^T + bias )
//   A, W bf16 row-major (lda / ldw); C f32 and/or Cbf bf16 outputs, ldc.
//   Register tiling: each wave computes a 32x32 output tile (2x2 wmma tiles),
//   so each A/B fragment is reused twice -> 2 b128 loads per wmma (vs 4 naive).
//   grid.x = N/256 (8 waves/block, one 32-wide tile each), grid.y = M/32.
// ---------------------------------------------------------------------------
__global__ __launch_bounds__(256)
void gemm_wmma_bf16(const bf16_t* __restrict__ A, int lda,
                    const bf16_t* __restrict__ W, int ldw,
                    float* __restrict__ C, bf16_t* __restrict__ Cbf, int ldc,
                    const float* __restrict__ addend,
                    const float* __restrict__ bias,
                    int K, int relu)
{
    int wave = threadIdx.x >> 5;
    int ncol = (blockIdx.x * 8 + wave) * 32;
    int mrow = blockIdx.y * 32;

    const bf16_t* A0 = A + (size_t)mrow * lda;
    const bf16_t* A1 = A0 + (size_t)16 * lda;
    const bf16_t* W0 = W + (size_t)ncol * ldw;
    const bf16_t* W1 = W0 + (size_t)16 * ldw;

    v8f acc00 = {}, acc01 = {}, acc10 = {}, acc11 = {};
    for (int kk = 0; kk < K; kk += 32) {
        // Prefetch next K-chunk of the streaming B operand (global_prefetch_b8).
        if (kk + 32 < K) {
            __builtin_prefetch(W0 + kk + 32, 0, 1);
            __builtin_prefetch(W1 + kk + 32, 0, 1);
        }
        v16bf a0 = load_frag16x32(A0 + kk, lda);
        v16bf a1 = load_frag16x32(A1 + kk, lda);
        v16bf b0 = load_frag16x32(W0 + kk, ldw);
        v16bf b1 = load_frag16x32(W1 + kk, ldw);
        acc00 = wmma_bf16(a0, b0, acc00);
        acc01 = wmma_bf16(a0, b1, acc01);
        acc10 = wmma_bf16(a1, b0, acc10);
        acc11 = wmma_bf16(a1, b1, acc11);
    }

    store_tile(acc00, mrow,      ncol,      C, Cbf, ldc, addend, bias, relu);
    store_tile(acc01, mrow,      ncol + 16, C, Cbf, ldc, addend, bias, relu);
    store_tile(acc10, mrow + 16, ncol,      C, Cbf, ldc, addend, bias, relu);
    store_tile(acc11, mrow + 16, ncol + 16, C, Cbf, ldc, addend, bias, relu);
}

// ---------------------------------------------------------------------------
// f32 -> bf16 conversion (grid-stride)
// ---------------------------------------------------------------------------
__global__ void cvt_f32_bf16(const float* __restrict__ in, bf16_t* __restrict__ out, size_t n) {
    size_t i = (size_t)blockIdx.x * blockDim.x + threadIdx.x;
    size_t s = (size_t)gridDim.x * blockDim.x;
    for (; i < n; i += s) out[i] = (bf16_t)in[i];
}

__global__ void add_vec2(const float* __restrict__ a, const float* __restrict__ b,
                         float* __restrict__ out, int n) {
    int i = blockIdx.x * blockDim.x + threadIdx.x;
    if (i < n) out[i] = a[i] + b[i];
}

// ---------------------------------------------------------------------------
// Embedding gather (+bf16 convert). Token n = s*Bz + b (time-major rows).
// Optional second copy into class_input rows (b*Tl + t), cols [0,512).
// ---------------------------------------------------------------------------
__global__ void gather_embed(const int* __restrict__ ids, const float* __restrict__ emb,
                             bf16_t* __restrict__ out, bf16_t* __restrict__ out2,
                             int seq_len)
{
    int n = blockIdx.x;                 // 0 .. seq_len*Bz-1
    int s = n / Bz, b = n % Bz;
    int id = ids[b * seq_len + s];
    const float* src = emb + (size_t)id * E;
    bf16_t* dst  = out + (size_t)n * E;
    bf16_t* dst2 = out2 ? out2 + ((size_t)b * Tl + s) * CIN : nullptr;
    for (int e = threadIdx.x; e < E; e += blockDim.x) {
        bf16_t v = (bf16_t)src[e];
        dst[e] = v;
        if (dst2) dst2[e] = v;
    }
}

__device__ __forceinline__ float sigf(float x) { return 1.0f / (1.0f + __expf(-x)); }

// ---------------------------------------------------------------------------
// Encoder LSTM cell (pointwise). g[B,4H] gate order i,f,g,o.
// Writes h (f32 + bf16) and enc_output[b][s_store][coloff + j] (f32 + bf16).
// ---------------------------------------------------------------------------
__global__ void lstm_cell_enc(const float* __restrict__ g,
                              float* __restrict__ h, float* __restrict__ c,
                              bf16_t* __restrict__ h_bf,
                              float* __restrict__ enc_f, bf16_t* __restrict__ enc_bf,
                              int s_store, int coloff)
{
    int idx = blockIdx.x * blockDim.x + threadIdx.x;   // [0, Bz*H)
    int b = idx / H, j = idx % H;
    const float* gr = g + (size_t)b * G4H;
    float iv = sigf(gr[j]);
    float fv = sigf(gr[H + j]);
    float gv = tanhf(gr[2 * H + j]);
    float ov = sigf(gr[3 * H + j]);
    float cc = fv * c[idx] + iv * gv;
    float hh = ov * tanhf(cc);
    c[idx] = cc;
    h[idx] = hh;
    h_bf[idx] = (bf16_t)hh;
    size_t eo = ((size_t)b * Sl + s_store) * H2 + coloff + j;
    enc_f[eo]  = hh;
    enc_bf[eo] = (bf16_t)hh;
}

// Decoder cell: also packs h into class_input rows (b*Tl + t), cols [1536,2048).
__global__ void lstm_cell_dec(const float* __restrict__ g,
                              float* __restrict__ h, float* __restrict__ c,
                              bf16_t* __restrict__ h_bf,
                              bf16_t* __restrict__ class_in, int t)
{
    int idx = blockIdx.x * blockDim.x + threadIdx.x;
    int b = idx / H, j = idx % H;
    const float* gr = g + (size_t)b * G4H;
    float iv = sigf(gr[j]);
    float fv = sigf(gr[H + j]);
    float gv = tanhf(gr[2 * H + j]);
    float ov = sigf(gr[3 * H + j]);
    float cc = fv * c[idx] + iv * gv;
    float hh = ov * tanhf(cc);
    c[idx] = cc;
    h[idx] = hh;
    h_bf[idx] = (bf16_t)hh;
    class_in[((size_t)b * Tl + t) * CIN + (E + H2) + j] = (bf16_t)hh;
}

// ---------------------------------------------------------------------------
// Attention scores: one wave per (b,s).
// score[b,s] = att2_b + sum_j att2_W[j] * relu(P_enc[b*S+s][j] + P_dec[b][j])
// ---------------------------------------------------------------------------
__global__ __launch_bounds__(256)
void att_score(const float* __restrict__ Penc, const float* __restrict__ Pdec,
               const float* __restrict__ att2W, const float* __restrict__ att2b,
               float* __restrict__ scores)
{
    int wave = threadIdx.x >> 5, lane = threadIdx.x & 31;
    int widx = blockIdx.x * 8 + wave;       // = b*Sl + s
    int b = widx / Sl;
    const float* pe = Penc + (size_t)widx * H3;
    const float* pd = Pdec + (size_t)b * H3;
    float acc = 0.0f;
    for (int j = lane; j < H3; j += 32) {
        float v = pe[j] + pd[j];
        if (v < 0.0f) v = 0.0f;
        acc += v * att2W[j];
    }
    #pragma unroll
    for (int off = 16; off; off >>= 1) acc += __shfl_xor(acc, off, 32);
    if (lane == 0) scores[widx] = acc + att2b[0];
}

// ---------------------------------------------------------------------------
// Softmax over S + weighted context. One block per batch row.
// ctx[b][j] = sum_s softmax(scores[b])[s] * enc_output[b][s][j]
// Writes ctx_bf[32,1024] and class_input cols [512,1536).
// ---------------------------------------------------------------------------
__global__ __launch_bounds__(256)
void att_softmax_ctx(const float* __restrict__ scores, const float* __restrict__ enc_f,
                     bf16_t* __restrict__ ctx_bf, bf16_t* __restrict__ class_in, int t)
{
    __shared__ float tmp[256];
    __shared__ float wsm[Sl];
    int b = blockIdx.x, tid = threadIdx.x;
    float sc = (tid < Sl) ? scores[b * Sl + tid] : -3.0e38f;
    tmp[tid] = sc; __syncthreads();
    for (int off = 128; off; off >>= 1) {
        if (tid < off) tmp[tid] = fmaxf(tmp[tid], tmp[tid + off]);
        __syncthreads();
    }
    float m = tmp[0]; __syncthreads();
    float e = (tid < Sl) ? __expf(sc - m) : 0.0f;
    tmp[tid] = e; __syncthreads();
    for (int off = 128; off; off >>= 1) {
        if (tid < off) tmp[tid] += tmp[tid + off];
        __syncthreads();
    }
    float inv = 1.0f / tmp[0];
    if (tid < Sl) wsm[tid] = e * inv;
    __syncthreads();

    const float* eb = enc_f + (size_t)b * Sl * H2;
    for (int j = tid; j < H2; j += 256) {
        float a = 0.0f;
        for (int s2 = 0; s2 < Sl; ++s2) a += wsm[s2] * eb[(size_t)s2 * H2 + j];
        bf16_t v = (bf16_t)a;
        ctx_bf[b * H2 + j] = v;
        class_in[((size_t)b * Tl + t) * CIN + E + j] = v;
    }
}

// ---------------------------------------------------------------------------
// Host side
// ---------------------------------------------------------------------------
static inline size_t al256(size_t x) { return (x + 255) & ~(size_t)255; }

static void gemm(hipStream_t st, const bf16_t* A, int lda, const bf16_t* W, int ldw,
                 float* C, bf16_t* Cbf, int ldc, const float* addend, const float* bias,
                 int M, int N, int K, int relu)
{
    dim3 grid(N / 256, M / 32);
    gemm_wmma_bf16<<<grid, 256, 0, st>>>(A, lda, W, ldw, C, Cbf, ldc, addend, bias, K, relu);
}

extern "C" void kernel_launch(void* const* d_in, const int* in_sizes, int n_in,
                              void* d_out, int out_size, void* d_ws, size_t ws_size,
                              hipStream_t stream)
{
    (void)in_sizes; (void)n_in; (void)out_size; (void)ws_size;

    const int*   src_ids  = (const int*)  d_in[0];
    const int*   tgt_ids  = (const int*)  d_in[1];
    const float* src_emb  = (const float*)d_in[2];
    const float* tgt_emb  = (const float*)d_in[3];
    const float* eWih_f   = (const float*)d_in[4];
    const float* eWhh_f   = (const float*)d_in[5];
    const float* ebih_f   = (const float*)d_in[6];
    const float* ebhh_f   = (const float*)d_in[7];
    const float* eWih_b   = (const float*)d_in[8];
    const float* eWhh_b   = (const float*)d_in[9];
    const float* ebih_b   = (const float*)d_in[10];
    const float* ebhh_b   = (const float*)d_in[11];
    const float* dWih     = (const float*)d_in[12];
    const float* dWhh     = (const float*)d_in[13];
    const float* dbih     = (const float*)d_in[14];
    const float* dbhh     = (const float*)d_in[15];
    const float* att1W    = (const float*)d_in[16];
    const float* att1b    = (const float*)d_in[17];
    const float* att2W    = (const float*)d_in[18];
    const float* att2b    = (const float*)d_in[19];
    const float* cls1W    = (const float*)d_in[20];
    const float* cls1b    = (const float*)d_in[21];
    const float* cls2W    = (const float*)d_in[22];
    const float* cls2b    = (const float*)d_in[23];
    float* out = (float*)d_out;

    // ---- workspace carve-out ----
    char* ws = (char*)d_ws; size_t off = 0;
    auto allocB = [&](size_t bytes) { void* p = ws + off; off = al256(off + bytes); return p; };

    bf16_t* wihF_bf = (bf16_t*)allocB((size_t)G4H * E  * 2);
    bf16_t* whhF_bf = (bf16_t*)allocB((size_t)G4H * H  * 2);
    bf16_t* wihB_bf = (bf16_t*)allocB((size_t)G4H * E  * 2);
    bf16_t* whhB_bf = (bf16_t*)allocB((size_t)G4H * H  * 2);
    bf16_t* dWih_bf = (bf16_t*)allocB((size_t)G4H * (E + H2) * 2);
    bf16_t* dWhh_bf = (bf16_t*)allocB((size_t)G4H * H  * 2);
    bf16_t* att1_bf = (bf16_t*)allocB((size_t)H3  * H3 * 2);
    bf16_t* cls1_bf = (bf16_t*)allocB((size_t)H2  * CIN * 2);
    bf16_t* cls2_bf = (bf16_t*)allocB((size_t)VT  * H2 * 2);

    float* biasF = (float*)allocB(G4H * 4);
    float* biasB = (float*)allocB(G4H * 4);
    float* biasD = (float*)allocB(G4H * 4);

    const int NTOK = Sl * Bz;                           // 3200
    bf16_t* x_bf    = (bf16_t*)allocB((size_t)NTOK * E * 2);
    bf16_t* temb_bf = (bf16_t*)allocB((size_t)NTOK * E * 2);
    float*  preF    = (float*) allocB((size_t)NTOK * G4H * 4);
    float*  preB    = (float*) allocB((size_t)NTOK * G4H * 4);
    float*  preD    = (float*) allocB((size_t)NTOK * G4H * 4);
    float*  enc_f   = (float*) allocB((size_t)NTOK * H2 * 4);
    bf16_t* enc_bf  = (bf16_t*)allocB((size_t)NTOK * H2 * 2);
    float*  Penc    = (float*) allocB((size_t)NTOK * H3 * 4);
    float*  Pdec    = (float*) allocB((size_t)Bz * H3 * 4);
    float*  scores  = (float*) allocB((size_t)NTOK * 4);
    bf16_t* ctx_bf  = (bf16_t*)allocB((size_t)Bz * H2 * 2);
    float*  gF      = (float*) allocB((size_t)Bz * G4H * 4);
    float*  gB      = (float*) allocB((size_t)Bz * G4H * 4);
    float*  gD      = (float*) allocB((size_t)Bz * G4H * 4);
    float*  hF      = (float*) allocB((size_t)Bz * H * 4);
    float*  cF      = (float*) allocB((size_t)Bz * H * 4);
    float*  hB      = (float*) allocB((size_t)Bz * H * 4);
    float*  cB      = (float*) allocB((size_t)Bz * H * 4);
    bf16_t* hF_bf   = (bf16_t*)allocB((size_t)Bz * H * 2);
    bf16_t* hB_bf   = (bf16_t*)allocB((size_t)Bz * H * 2);
    bf16_t* clsin   = (bf16_t*)allocB((size_t)NTOK * CIN * 2);
    bf16_t* c1_bf   = (bf16_t*)allocB((size_t)NTOK * H2 * 2);

    // ---- weight conversions f32 -> bf16 ----
    auto cvt = [&](const float* in, bf16_t* o, size_t n) {
        int blocks = (int)((n + 255) / 256); if (blocks > 2048) blocks = 2048;
        cvt_f32_bf16<<<blocks, 256, 0, stream>>>(in, o, n);
    };
    cvt(eWih_f, wihF_bf, (size_t)G4H * E);
    cvt(eWhh_f, whhF_bf, (size_t)G4H * H);
    cvt(eWih_b, wihB_bf, (size_t)G4H * E);
    cvt(eWhh_b, whhB_bf, (size_t)G4H * H);
    cvt(dWih,   dWih_bf, (size_t)G4H * (E + H2));
    cvt(dWhh,   dWhh_bf, (size_t)G4H * H);
    cvt(att1W,  att1_bf, (size_t)H3 * H3);
    cvt(cls1W,  cls1_bf, (size_t)H2 * CIN);
    cvt(cls2W,  cls2_bf, (size_t)VT * H2);

    add_vec2<<<G4H / 256, 256, 0, stream>>>(ebih_f, ebhh_f, biasF, G4H);
    add_vec2<<<G4H / 256, 256, 0, stream>>>(ebih_b, ebhh_b, biasB, G4H);
    add_vec2<<<G4H / 256, 256, 0, stream>>>(dbih,   dbhh,   biasD, G4H);

    // ---- embedding gathers (time-major rows s*Bz+b); target also packs class_input[:, 0:512) ----
    gather_embed<<<NTOK, 256, 0, stream>>>(src_ids, src_emb, x_bf,    nullptr, Sl);
    gather_embed<<<NTOK, 256, 0, stream>>>(tgt_ids, tgt_emb, temb_bf, clsin,   Tl);

    // ---- precompute all input projections (big WMMA GEMMs) ----
    // preF/preB[s*B+b, 4H] = x @ Wih^T + (bih+bhh)
    gemm(stream, x_bf, E, wihF_bf, E, preF, nullptr, G4H, nullptr, biasF, NTOK, G4H, E, 0);
    gemm(stream, x_bf, E, wihB_bf, E, preB, nullptr, G4H, nullptr, biasB, NTOK, G4H, E, 0);
    // preD[t*B+b, 4H] = temb @ dec_Wih[:, :E]^T + (bih+bhh)
    gemm(stream, temb_bf, E, dWih_bf, E + H2, preD, nullptr, G4H, nullptr, biasD, NTOK, G4H, E, 0);

    // ---- encoder recurrence ----
    hipMemsetAsync(hF, 0, (size_t)Bz * H * 4, stream);
    hipMemsetAsync(cF, 0, (size_t)Bz * H * 4, stream);
    hipMemsetAsync(hB, 0, (size_t)Bz * H * 4, stream);
    hipMemsetAsync(cB, 0, (size_t)Bz * H * 4, stream);
    hipMemsetAsync(hF_bf, 0, (size_t)Bz * H * 2, stream);
    hipMemsetAsync(hB_bf, 0, (size_t)Bz * H * 2, stream);

    const int cellBlocks = (Bz * H) / 256;   // 64
    for (int t = 0; t < Sl; ++t) {
        // forward: g = preF[t] + h @ Whh^T
        gemm(stream, hF_bf, H, whhF_bf, H, gF, nullptr, G4H,
             preF + (size_t)t * Bz * G4H, nullptr, Bz, G4H, H, 0);
        lstm_cell_enc<<<cellBlocks, 256, 0, stream>>>(gF, hF, cF, hF_bf, enc_f, enc_bf, t, 0);
        // backward processes x[Sl-1-t], stores into enc_output[:, Sl-1-t, H:2H]
        int s = Sl - 1 - t;
        gemm(stream, hB_bf, H, whhB_bf, H, gB, nullptr, G4H,
             preB + (size_t)s * Bz * G4H, nullptr, Bz, G4H, H, 0);
        lstm_cell_enc<<<cellBlocks, 256, 0, stream>>>(gB, hB, cB, hB_bf, enc_f, enc_bf, s, H);
    }

    // ---- factorized attention: P_enc = enc_output @ att1_W[:, :2H]^T + att1_b (once) ----
    gemm(stream, enc_bf, H2, att1_bf, H3, Penc, nullptr, H3, nullptr, att1b, NTOK, H3, H2, 0);

    // ---- decoder recurrence (init state = final forward encoder state: reuse hF/cF) ----
    for (int t = 0; t < Tl; ++t) {
        // P_dec = h @ att1_W[:, 2H:3H]^T
        gemm(stream, hF_bf, H, att1_bf + H2, H3, Pdec, nullptr, H3, nullptr, nullptr, Bz, H3, H, 0);
        att_score<<<NTOK / 8, 256, 0, stream>>>(Penc, Pdec, att2W, att2b, scores);
        att_softmax_ctx<<<Bz, 256, 0, stream>>>(scores, enc_f, ctx_bf, clsin, t);
        // g = preD[t] + ctx @ dec_Wih[:, E:]^T + h @ dec_Whh^T
        gemm(stream, ctx_bf, H2, dWih_bf + E, E + H2, gD, nullptr, G4H,
             preD + (size_t)t * Bz * G4H, nullptr, Bz, G4H, H2, 0);
        gemm(stream, hF_bf, H, dWhh_bf, H, gD, nullptr, G4H, gD, nullptr, Bz, G4H, H, 0);
        lstm_cell_dec<<<cellBlocks, 256, 0, stream>>>(gD, hF, cF, hF_bf, clsin, t);
    }

    // ---- classifier ----
    // c1 = relu(class_input @ cls1_W^T + b1) -> bf16
    gemm(stream, clsin, CIN, cls1_bf, CIN, nullptr, c1_bf, H2, nullptr, cls1b, NTOK, H2, CIN, 1);
    // out[b*T+t, VT] = c1 @ cls2_W^T + b2   (210 GFLOP, the big WMMA GEMM)
    gemm(stream, c1_bf, H2, cls2_bf, H2, out, nullptr, VT, nullptr, cls2b, NTOK, VT, H2, 0);
}
